// BasicGNN_82454782148695
// MI455X (gfx1250) — matrix operand order repaired and checked by reading, not compiled, not run on hardware
//
#include <hip/hip_runtime.h>
#include <hip/hip_bf16.h>

// ---------------------------------------------------------------------------
// BasicGNN on gfx1250 (MI455X): bf16 WMMA GEMMs + f32-atomic scatter-mean.
// Inputs (setup_inputs order):
//  0 x[N,128] f32          1 edge_index[2,E] i64
//  2 W1[128,128] 3 b1[128] 4 g1 5 be1 6 m1 7 v1
//  8 W2[128,128] 9 b2      10 g2 11 be2 12 m2 13 v2
// 14 Wc1[128,64] 15 bc1[64] 16 Wc2[64,2] 17 bc2[2]
// Output: logits [N,2] f32
// ---------------------------------------------------------------------------

typedef __attribute__((ext_vector_type(16))) __bf16 bf16x16;
typedef __attribute__((ext_vector_type(8)))  __bf16 bf16x8;
typedef __attribute__((ext_vector_type(4)))  __bf16 bf16x4;
typedef __attribute__((ext_vector_type(8)))  float  v8f;

#define HIDK 128   // K dimension of every GEMM (DIN == HID == 128)
#define EPSV 1e-5f

// ---- prep: W[k][c] (f32, row-major, K=128) -> Wt[c*128+k] (bf16, col-major)
__global__ void k_transpose_bf16(const float* __restrict__ W,
                                 __bf16* __restrict__ Wt, int ncol) {
    int i = blockIdx.x * blockDim.x + threadIdx.x;
    if (i >= HIDK * ncol) return;
    int k = i / ncol, c = i % ncol;
    Wt[c * HIDK + k] = (__bf16)W[i];
}

// ---- prep: fold BN into affine form: y = val*scale[f] + shift[f]
__global__ void k_bn_prep(const float* __restrict__ g, const float* __restrict__ be,
                          const float* __restrict__ m, const float* __restrict__ v,
                          float* __restrict__ scale, float* __restrict__ shift) {
    int f = threadIdx.x;
    if (f < HIDK) {
        float s = g[f] * rsqrtf(v[f] + EPSV);
        scale[f] = s;
        shift[f] = be[f] - m[f] * s;
    }
}

// ---- GEMM: Out[N,NCOL] = X[N,128] @ W + bias  (optional ReLU)
// One block = 16 rows; NCOL/16 waves; wave w owns columns [16w,16w+16).
// Fragments follow CDNA5 16-bit WMMA layouts (ISA 7.12.2):
//  A lane m|m+16 -> row m, K chunks {0-7,16-23} / {8-15,24-31}
//  B lane n|n+16 -> col n, K chunks {0-7,16-23} / {8-15,24-31} (col-major Wt)
//  C vgpr r, lanes 0-15 -> M=r; lanes 16-31 -> M=r+8
template <int NCOL, bool RELU>
__global__ __launch_bounds__(NCOL * 2)
void k_gemm_wmma(const float* __restrict__ X, const __bf16* __restrict__ Wt,
                 const float* __restrict__ bias, float* __restrict__ Out, int N) {
    __shared__ __align__(32) __bf16 tileA[16 * HIDK];   // 4 KB

    const int rowBase = blockIdx.x * 16;
    const bool fullTile = (rowBase + 16 <= N);

    // stage & convert the 16x128 f32 A-tile into LDS as bf16 (float4 granules)
    for (int c = threadIdx.x; c < 16 * HIDK / 4; c += NCOL * 2) {
        int r = c >> 5;                 // 32 float4 per row
        int k4 = (c & 31) * 4;
        int gr = rowBase + r;
        float4 xv = make_float4(0.f, 0.f, 0.f, 0.f);
        if (fullTile || gr < N)
            xv = *(const float4*)(X + (size_t)gr * HIDK + k4);
        bf16x4 bq = { (__bf16)xv.x, (__bf16)xv.y, (__bf16)xv.z, (__bf16)xv.w };
        *(bf16x4*)(tileA + r * HIDK + k4) = bq;
    }
    __syncthreads();

    const int wave = threadIdx.x >> 5;
    const int lane = threadIdx.x & 31;
    const int colBase = wave * 16;
    const int n  = lane & 15;
    const int hi = lane >> 4;          // half-wave select

    v8f acc = {};
    #pragma unroll
    for (int kb = 0; kb < HIDK / 32; ++kb) {
        const int bK = kb * 32;
        union { bf16x16 v; bf16x8 h[2]; } a;
        a.h[0] = *(const bf16x8*)(tileA + n * HIDK + bK + hi * 8);
        a.h[1] = *(const bf16x8*)(tileA + n * HIDK + bK + 16 + hi * 8);
        union { bf16x16 v; bf16x8 h[2]; } bu;
        bu.h[0] = *(const bf16x8*)(Wt + (size_t)(colBase + n) * HIDK + bK + hi * 8);
        bu.h[1] = *(const bf16x8*)(Wt + (size_t)(colBase + n) * HIDK + bK + 16 + hi * 8);
        bf16x16 b = bu.v;
        acc = __builtin_amdgcn_wmma_f32_16x16x32_bf16(
                  false, a.v, false, b, (short)0, acc, false, false);
    }

    const float bv = bias[colBase + n];
    float* outp = Out + ((size_t)(rowBase + hi * 8) * NCOL) + colBase + n;
    if (fullTile) {
        #pragma unroll
        for (int r = 0; r < 8; ++r) {
            float val = acc[r] + bv;
            if (RELU) val = fmaxf(val, 0.0f);
            outp[r * NCOL] = val;               // immediate-offset stores
        }
    } else {
        #pragma unroll
        for (int r = 0; r < 8; ++r) {
            int row = rowBase + hi * 8 + r;
            if (row < N) {
                float val = acc[r] + bv;
                if (RELU) val = fmaxf(val, 0.0f);
                outp[r * NCOL] = val;
            }
        }
    }
}

// ---- degree: init to 1 (self loop), count edges, then invert
__global__ void k_cnt_init(float* __restrict__ cnt, int N) {
    int i = blockIdx.x * blockDim.x + threadIdx.x;
    if (i < N) cnt[i] = 1.0f;
}
__global__ void k_edge_count(const long long* __restrict__ ei,
                             float* __restrict__ cnt, long long E) {
    long long e = (long long)blockIdx.x * blockDim.x + threadIdx.x;
    if (e < E) atomicAdd(&cnt[ei[E + e]], 1.0f);
}
__global__ void k_cnt_invert(float* __restrict__ cnt, int N) {
    int i = blockIdx.x * blockDim.x + threadIdx.x;
    if (i < N) cnt[i] = 1.0f / cnt[i];
}

// ---- scatter: AGG[dst] += H[src]; one thread per (edge, float4 chunk).
// AGG pre-initialized to H (self loops) via D2D memcpy. Per-wave the 32
// threads of an edge read one coalesced 512B row and issue 128 f32 atomics.
__global__ void k_edge_scatter(const long long* __restrict__ ei,
                               const float* __restrict__ H,
                               float* __restrict__ AGG, long long E) {
    long long t = (long long)blockIdx.x * blockDim.x + threadIdx.x;
    if (t >= E * 32) return;
    long long e = t >> 5;
    int c = (int)(t & 31) * 4;
    long long src = ei[e];
    long long dst = ei[E + e];
    float4 v = *(const float4*)(H + src * HIDK + c);
    float* o = AGG + dst * HIDK + c;
    atomicAdd(o + 0, v.x);
    atomicAdd(o + 1, v.y);
    atomicAdd(o + 2, v.z);
    atomicAdd(o + 3, v.w);
}

// ---- fused mean + batchnorm(affine) + relu
__global__ void k_mean_bn_relu(const float* __restrict__ AGG,
                               const float* __restrict__ invcnt,
                               const float* __restrict__ scale,
                               const float* __restrict__ shift,
                               float* __restrict__ out, long long total) {
    long long i = (long long)blockIdx.x * blockDim.x + threadIdx.x;
    if (i >= total) return;
    int f = (int)(i & (HIDK - 1));
    long long node = i >> 7;
    float val = AGG[i] * invcnt[node];
    out[i] = fmaxf(fmaf(val, scale[f], shift[f]), 0.0f);
}

// ---- final 64 -> 2 projection
__global__ void k_classifier_out(const float* __restrict__ Hd,
                                 const float* __restrict__ Wc2,
                                 const float* __restrict__ bc2,
                                 float* __restrict__ out, int N) {
    int i = blockIdx.x * blockDim.x + threadIdx.x;
    if (i >= N) return;
    const float* h = Hd + (size_t)i * 64;
    float o0 = bc2[0], o1 = bc2[1];
    #pragma unroll 4
    for (int j = 0; j < 64; ++j) {
        float x = h[j];
        o0 = fmaf(x, Wc2[j * 2 + 0], o0);
        o1 = fmaf(x, Wc2[j * 2 + 1], o1);
    }
    out[(size_t)i * 2 + 0] = o0;
    out[(size_t)i * 2 + 1] = o1;
}

static inline size_t align256(size_t x) { return (x + 255) & ~(size_t)255; }

extern "C" void kernel_launch(void* const* d_in, const int* in_sizes, int n_in,
                              void* d_out, int out_size, void* d_ws, size_t ws_size,
                              hipStream_t stream) {
    const int       N = in_sizes[0] / HIDK;        // 100000
    const long long E = (long long)in_sizes[1] / 2;

    const float*     x  = (const float*)d_in[0];
    const long long* ei = (const long long*)d_in[1];   // int64 edge_index
    const float *W1 = (const float*)d_in[2],  *b1  = (const float*)d_in[3];
    const float *g1 = (const float*)d_in[4],  *be1 = (const float*)d_in[5];
    const float *m1 = (const float*)d_in[6],  *v1  = (const float*)d_in[7];
    const float *W2 = (const float*)d_in[8],  *b2  = (const float*)d_in[9];
    const float *g2 = (const float*)d_in[10], *be2 = (const float*)d_in[11];
    const float *m2 = (const float*)d_in[12], *v2  = (const float*)d_in[13];
    const float *Wc1 = (const float*)d_in[14], *bc1 = (const float*)d_in[15];
    const float *Wc2 = (const float*)d_in[16], *bc2 = (const float*)d_in[17];
    float* out = (float*)d_out;

    // workspace carve-up (~103 MB; both feature buffers are L2-resident)
    char* ws = (char*)d_ws;
    size_t off = 0;
    const size_t featBytes = (size_t)N * HIDK * sizeof(float);
    float*  bufA = (float*)(ws + off); off += align256(featBytes);
    float*  bufB = (float*)(ws + off); off += align256(featBytes);
    float*  cnt  = (float*)(ws + off); off += align256((size_t)N * sizeof(float));
    __bf16* Wt1  = (__bf16*)(ws + off); off += align256(HIDK * HIDK * sizeof(__bf16));
    __bf16* Wt2  = (__bf16*)(ws + off); off += align256(HIDK * HIDK * sizeof(__bf16));
    __bf16* Wtc1 = (__bf16*)(ws + off); off += align256(HIDK * 64 * sizeof(__bf16));
    float*  sc1  = (float*)(ws + off); off += align256(HIDK * sizeof(float));
    float*  sh1  = (float*)(ws + off); off += align256(HIDK * sizeof(float));
    float*  sc2  = (float*)(ws + off); off += align256(HIDK * sizeof(float));
    float*  sh2  = (float*)(ws + off); off += align256(HIDK * sizeof(float));

    const int rowBlocks = (N + 15) / 16;

    // --- tiny prep: weights -> bf16 col-major, BN -> affine, degrees ---
    k_transpose_bf16<<<(HIDK * HIDK + 255) / 256, 256, 0, stream>>>(W1, Wt1, HIDK);
    k_transpose_bf16<<<(HIDK * HIDK + 255) / 256, 256, 0, stream>>>(W2, Wt2, HIDK);
    k_transpose_bf16<<<(HIDK * 64   + 255) / 256, 256, 0, stream>>>(Wc1, Wtc1, 64);
    k_bn_prep<<<1, HIDK, 0, stream>>>(g1, be1, m1, v1, sc1, sh1);
    k_bn_prep<<<1, HIDK, 0, stream>>>(g2, be2, m2, v2, sc2, sh2);
    k_cnt_init<<<(N + 255) / 256, 256, 0, stream>>>(cnt, N);
    k_edge_count<<<(int)((E + 255) / 256), 256, 0, stream>>>(ei, cnt, E);
    k_cnt_invert<<<(N + 255) / 256, 256, 0, stream>>>(cnt, N);

    const int scatterBlocks = (int)((E * 32 + 255) / 256);
    const long long total = (long long)N * HIDK;
    const int elemBlocks = (int)((total + 255) / 256);

    // --- layer 1: H1 = X@W1+b1 -> agg -> mean+BN+ReLU ---
    k_gemm_wmma<HIDK, false><<<rowBlocks, 256, 0, stream>>>(x, Wt1, b1, bufA, N);
    hipMemcpyAsync(bufB, bufA, featBytes, hipMemcpyDeviceToDevice, stream); // self loops
    k_edge_scatter<<<scatterBlocks, 256, 0, stream>>>(ei, bufA, bufB, E);
    k_mean_bn_relu<<<elemBlocks, 256, 0, stream>>>(bufB, cnt, sc1, sh1, bufA, total);

    // --- layer 2: H2 = X2@W2+b2 -> agg -> mean+BN+ReLU ---
    k_gemm_wmma<HIDK, false><<<rowBlocks, 256, 0, stream>>>(bufA, Wt2, b2, bufB, N);
    hipMemcpyAsync(bufA, bufB, featBytes, hipMemcpyDeviceToDevice, stream); // self loops
    k_edge_scatter<<<scatterBlocks, 256, 0, stream>>>(ei, bufB, bufA, E);
    k_mean_bn_relu<<<elemBlocks, 256, 0, stream>>>(bufA, cnt, sc2, sh2, bufB, total);

    // --- classifier: hidden = relu(X3@Wc1+bc1) [N,64]; out = hidden@Wc2+bc2 ---
    k_gemm_wmma<64, true><<<rowBlocks, 128, 0, stream>>>(bufB, Wtc1, bc1, bufA, N);
    k_classifier_out<<<(N + 255) / 256, 256, 0, stream>>>(bufA, Wc2, bc2, out, N);
}
